// EpipolarCrossViewAttention_84576495993156
// MI455X (gfx1250) — matrix operand order repaired and checked by reading, not compiled, not run on hardware
//
#include <hip/hip_runtime.h>

#define B_   4
#define C_   1024
#define HW_  2304          // 48*48 tokens per batch
#define D_   256
#define M_   (B_*HW_)      // 9216 total tokens
#define EPS_ 1e-6f
#define INV_TEMP_ 10.0f    // 1/GEO_TEMP
#define SCALE_ 0.0625f     // 256^-0.5
#define LN_EPS_ 1e-5f
#define KC_  128           // K-chunk staged per TDM transfer

typedef unsigned short u16;
typedef __attribute__((ext_vector_type(16))) __bf16 v16bf;
typedef __attribute__((ext_vector_type(8)))  float  v8f;
typedef __attribute__((ext_vector_type(4)))  unsigned int v4u;
typedef __attribute__((ext_vector_type(8)))  unsigned int v8u;

__device__ __forceinline__ u16 f2bf(float f) {
  unsigned u = __builtin_bit_cast(unsigned, f);
  unsigned r = u + 0x7fffu + ((u >> 16) & 1u);   // round-to-nearest-even
  return (u16)(r >> 16);
}
__device__ __forceinline__ __bf16 bfbits(u16 x) {
  return __builtin_bit_cast(__bf16, x);
}

// ---------------------------------------------------------------- weights -> bf16
__global__ void k_cvt_bf16(const float* __restrict__ src, u16* __restrict__ dst, int n) {
  for (int i = blockIdx.x * blockDim.x + threadIdx.x; i < n; i += gridDim.x * blockDim.x)
    dst[i] = f2bf(src[i]);
}

// ---------------------------------------------------------------- LayerNorm + token matrices (bf16)
__global__ __launch_bounds__(256) void k_prep(
    const float* __restrict__ qmap, const float* __restrict__ kmap,
    const float* __restrict__ gq, const float* __restrict__ bq,
    const float* __restrict__ gk, const float* __restrict__ bk,
    u16* __restrict__ qn, u16* __restrict__ kn, u16* __restrict__ vt)
{
  const int blocksPerB = HW_ / 32;
  int b   = blockIdx.x / blocksPerB;
  int n0  = (blockIdx.x % blocksPerB) * 32;
  int tok = threadIdx.x & 31;
  int part = threadIdx.x >> 5;          // 8 parts * 128 channels = 1024
  const float* qb = qmap + (size_t)b * C_ * HW_;
  const float* kb_ = kmap + (size_t)b * C_ * HW_;

  float sq = 0.f, sqq = 0.f, sk = 0.f, skk = 0.f;
  int c0 = part * 128;
  for (int c = c0; c < c0 + 128; ++c) {
    float x = qb[(size_t)c * HW_ + n0 + tok]; sq += x; sqq += x * x;
    float y = kb_[(size_t)c * HW_ + n0 + tok]; sk += y; skk += y * y;
  }
  __shared__ float rs[4][8][32];
  rs[0][part][tok] = sq;  rs[1][part][tok] = sqq;
  rs[2][part][tok] = sk;  rs[3][part][tok] = skk;
  __syncthreads();
  __shared__ float muq[32], rq[32], muk[32], rk[32];
  if (part == 0) {
    float a = 0.f, bb = 0.f, cc = 0.f, dd = 0.f;
    for (int p = 0; p < 8; ++p) { a += rs[0][p][tok]; bb += rs[1][p][tok]; cc += rs[2][p][tok]; dd += rs[3][p][tok]; }
    float mu = a / (float)C_, var = bb / (float)C_ - mu * mu;
    muq[tok] = mu; rq[tok] = rsqrtf(var + LN_EPS_);
    mu = cc / (float)C_; var = dd / (float)C_ - mu * mu;
    muk[tok] = mu; rk[tok] = rsqrtf(var + LN_EPS_);
  }
  __syncthreads();
  float mQ = muq[tok], rQ = rq[tok], mK = muk[tok], rK = rk[tok];
  size_t orow = ((size_t)b * HW_ + n0 + tok) * C_;
  for (int c = c0; c < c0 + 128; ++c) {
    float x = qb[(size_t)c * HW_ + n0 + tok];
    float y = kb_[(size_t)c * HW_ + n0 + tok];
    qn[orow + c] = f2bf((x - mQ) * rQ * gq[c] + bq[c]);
    kn[orow + c] = f2bf((y - mK) * rK * gk[c] + bk[c]);
    vt[orow + c] = f2bf(y);
  }
}

// ---------------------------------------------------------------- ray geometry precompute
__global__ void k_rays(const float* __restrict__ pq, const float* __restrict__ pk,
                       float* __restrict__ dq, float* __restrict__ mq, float* __restrict__ nq,
                       float* __restrict__ dk, float* __restrict__ mk, float* __restrict__ nk)
{
  int i = blockIdx.x * blockDim.x + threadIdx.x;
  if (i >= M_) return;
  int b = i / HW_, n = i - b * HW_;
  {
    const float* P = pq + (size_t)b * 6 * HW_;
    float d0 = P[0 * HW_ + n], d1 = P[1 * HW_ + n], d2 = P[2 * HW_ + n];
    float m0 = P[3 * HW_ + n], m1 = P[4 * HW_ + n], m2 = P[5 * HW_ + n];
    float dn = fmaxf(sqrtf(d0 * d0 + d1 * d1 + d2 * d2), EPS_);
    dq[i * 3 + 0] = d0 / dn; dq[i * 3 + 1] = d1 / dn; dq[i * 3 + 2] = d2 / dn;
    mq[i * 3 + 0] = m0; mq[i * 3 + 1] = m1; mq[i * 3 + 2] = m2;
    nq[i] = sqrtf(m0 * m0 + m1 * m1 + m2 * m2);
  }
  {
    const float* P = pk + (size_t)b * 6 * HW_;
    float d0 = P[0 * HW_ + n], d1 = P[1 * HW_ + n], d2 = P[2 * HW_ + n];
    float m0 = P[3 * HW_ + n], m1 = P[4 * HW_ + n], m2 = P[5 * HW_ + n];
    float dn = fmaxf(sqrtf(d0 * d0 + d1 * d1 + d2 * d2), EPS_);
    dk[i * 3 + 0] = d0 / dn; dk[i * 3 + 1] = d1 / dn; dk[i * 3 + 2] = d2 / dn;
    mk[i * 3 + 0] = m0; mk[i * 3 + 1] = m1; mk[i * 3 + 2] = m2;
    nk[i] = sqrtf(m0 * m0 + m1 * m1 + m2 * m2);
  }
}

// ---------------------------------------------------------------- bf16 WMMA GEMM with TDM-staged B tile
// C[M x Nout] = A[M x K](bf16) * Bw[Nout x K]^T (bf16) + bias.
// Block = 8 waves; each wave owns a 16(M) x 64(N) tile. The 64-column weight tile is
// shared by all 8 waves, so it is DMA'd into LDS by the Tensor Data Mover
// (tensor_load_to_lds, double-buffered, TENSORcnt-tracked) while waves compute on the
// previous chunk. A fragments stream from global with prefetch (no reuse to exploit).
__global__ __launch_bounds__(256) void k_gemm_wmma(
    const u16* __restrict__ A, const u16* __restrict__ Bw,
    const float* __restrict__ bias, float* __restrict__ Cout,
    int K, int Nout, int transOut)
{
  __shared__ u16 ldsB[2][64][KC_];      // 2 x 16 KB double buffer
  int lane = threadIdx.x & 31;
  int wave = threadIdx.x >> 5;
  int m0 = (blockIdx.x * 8 + wave) * 16;
  int n0 = blockIdx.y * 64;
  int mrow = m0 + (lane & 15);
  int kh = lane >> 4;                   // K-half select per ISA A/B lane layout
  int nchunks = K / KC_;

  unsigned long long gbase = (unsigned long long)(uintptr_t)(Bw + (size_t)n0 * K);
  v4u g2, g3;
#pragma unroll
  for (int i = 0; i < 4; ++i) { g2[i] = 0u; g3[i] = 0u; }

  auto issue_tdm = [&](int chunk) {
    unsigned long long ga = gbase + (unsigned long long)chunk * (KC_ * 2);
    unsigned ldsaddr = (unsigned)(uintptr_t)&ldsB[chunk & 1][0][0];
    v4u g0; v8u g1;
    // D# group 0: count=1 | lds_addr | global_addr | type=2 ("image")
    g0[0] = 1u;
    g0[1] = ldsaddr;
    g0[2] = (unsigned)ga;
    g0[3] = (unsigned)(ga >> 32) | (2u << 30);
    // D# group 1: 2D tile, data_size=2B, tile = KC_ x 64, row stride = K elements
    g1[0] = 1u << 16;                                       // data_size = 1 (2 bytes)
    g1[1] = ((unsigned)K & 0xFFFFu) << 16;                  // tensor_dim0[15:0]
    g1[2] = (((unsigned)K >> 16) & 0xFFFFu) | (64u << 16);  // tensor_dim0[31:16] | tensor_dim1[15:0]
    g1[3] = ((unsigned)KC_) << 16;                          // tensor_dim1[31:16]=0 | tile_dim0
    g1[4] = 64u;                                            // tile_dim1=64, tile_dim2=0
    g1[5] = (unsigned)K;                                    // tensor_dim0_stride[31:0]
    g1[6] = 0u;                                             // stride hi, dim1_stride lo
    g1[7] = 0u;
    asm volatile("tensor_load_to_lds %0, %1, %2, %3"
                 :: "s"(g0), "s"(g1), "s"(g2), "s"(g3) : "memory");
  };

  if (wave == 0) issue_tdm(0);

  v8f acc[4] = {};

  for (int c = 0; c < nchunks; ++c) {
    if (wave == 0) {
      if (c + 1 < nchunks) {
        issue_tdm(c + 1);                       // overlap next chunk with compute
        __builtin_amdgcn_s_wait_tensorcnt(1);   // chunk c landed; c+1 in flight
      } else {
        __builtin_amdgcn_s_wait_tensorcnt(0);
      }
    }
    __syncthreads();
    int bufp = c & 1;
    for (int kk0 = 0; kk0 < KC_; kk0 += 32) {
      int kb = c * KC_ + kk0;
      const u16* pa = A + (size_t)mrow * K + kb + kh * 8;
      __builtin_prefetch(pa + 64, 0, 1);        // global_prefetch_b8, two K steps ahead
      v16bf af;
#pragma unroll
      for (int i = 0; i < 8; ++i) af[i]     = bfbits(pa[i]);
#pragma unroll
      for (int i = 0; i < 8; ++i) af[8 + i] = bfbits(pa[16 + i]);
#pragma unroll
      for (int nt = 0; nt < 4; ++nt) {
        const u16* pb = &ldsB[bufp][nt * 16 + (lane & 15)][kk0 + kh * 8];
        v16bf bf;
#pragma unroll
        for (int i = 0; i < 8; ++i) { bf[i] = bfbits(pb[i]); bf[8 + i] = bfbits(pb[16 + i]); }
        acc[nt] = __builtin_amdgcn_wmma_f32_16x16x32_bf16(
            false, af, false, bf, (short)0, acc[nt], false, false);
      }
    }
    __syncthreads();                            // buffer free before TDM overwrites it
  }

#pragma unroll
  for (int nt = 0; nt < 4; ++nt) {
    int col = n0 + nt * 16 + (lane & 15);
    float bv = bias[col];
#pragma unroll
    for (int r = 0; r < 8; ++r) {
      int row = m0 + r + kh * 8;                // C/D layout: VGPR r -> M=r (+8 for lanes 16-31)
      float val = acc[nt][r] + bv;
      if (!transOut) {
        Cout[(size_t)row * Nout + col] = val;
      } else {
        int bb = row / HW_, tok = row - bb * HW_;
        Cout[((size_t)bb * C_ + col) * HW_ + tok] = val;
      }
    }
  }
}

// ---------------------------------------------------------------- top-32 epipolar attention
// One wave per query. -10000 mask => softmax is exactly over the 32 kept keys in fp32.
__global__ __launch_bounds__(256) void k_attn_topk(
    const float* __restrict__ q, const float* __restrict__ k, const float* __restrict__ v,
    const float* __restrict__ dq, const float* __restrict__ mq, const float* __restrict__ nq,
    const float* __restrict__ dk, const float* __restrict__ mk, const float* __restrict__ nk,
    u16* __restrict__ fused)
{
  const float FINF = 3.4e38f;
  int lane = threadIdx.x & 31;
  int wave = threadIdx.x >> 5;
  int qi = blockIdx.x * 8 + wave;
  int b = qi / HW_;

  float dq0 = dq[qi * 3 + 0], dq1 = dq[qi * 3 + 1], dq2 = dq[qi * 3 + 2];
  float mq0 = mq[qi * 3 + 0], mq1 = mq[qi * 3 + 1], mq2 = mq[qi * 3 + 2];
  float nqi = nq[qi];
  const float* mkb = mk + (size_t)b * HW_ * 3;
  const float* dkb = dk + (size_t)b * HW_ * 3;
  const float* nkb = nk + (size_t)b * HW_;

  // per-lane top-32 (replace-max)
  float best[32]; int bidx[32];
#pragma unroll
  for (int t = 0; t < 32; ++t) { best[t] = FINF; bidx[t] = 0; }
  float curMax = FINF; int maxPos = 0;
  for (int j = lane; j < HW_; j += 32) {
    float rec = fabsf(dq0 * mkb[j * 3] + dq1 * mkb[j * 3 + 1] + dq2 * mkb[j * 3 + 2]
                    + dkb[j * 3] * mq0 + dkb[j * 3 + 1] * mq1 + dkb[j * 3 + 2] * mq2)
                / (nqi + nkb[j] + EPS_);
    if (rec < curMax) {
      best[maxPos] = rec; bidx[maxPos] = j;
      curMax = -FINF;
      for (int t = 0; t < 32; ++t) if (best[t] > curMax) { curMax = best[t]; maxPos = t; }
    }
  }
  // lane-local ascending insertion sort
  for (int a2 = 1; a2 < 32; ++a2) {
    float v0 = best[a2]; int i0 = bidx[a2]; int p = a2 - 1;
    while (p >= 0 && best[p] > v0) { best[p + 1] = best[p]; bidx[p + 1] = bidx[p]; --p; }
    best[p + 1] = v0; bidx[p + 1] = i0;
  }
  // wave merge: lane t ends up holding the t-th smallest (value, index)
  int keepIdx = 0; float keepRec = 0.f; int hp = 0;
  for (int t = 0; t < 32; ++t) {
    float cand   = (hp < 32) ? best[hp] : FINF;
    int   candId = (hp < 32) ? bidx[hp] : 0;
    float mval = cand; int mlane = lane;
    for (int off = 16; off > 0; off >>= 1) {
      float ov = __shfl_xor(mval, off);
      int   ol = __shfl_xor(mlane, off);
      if (ov < mval || (ov == mval && ol < mlane)) { mval = ov; mlane = ol; }
    }
    int widx = __shfl(candId, mlane);
    if (lane == t) { keepIdx = widx; keepRec = mval; }
    if (lane == mlane) ++hp;
  }

  // attention over the 32 kept keys (cooperative across the wave, coalesced)
  const float* qrow = q + (size_t)qi * D_;
  float qreg[8];
#pragma unroll
  for (int c = 0; c < 8; ++c) qreg[c] = qrow[lane + 32 * c];

  float logit = 0.f;
  for (int t = 0; t < 32; ++t) {
    int j = __shfl(keepIdx, t);
    float rec = __shfl(keepRec, t);
    const float* kp = k + ((size_t)b * HW_ + j) * D_;
    float p = 0.f;
#pragma unroll
    for (int c = 0; c < 8; ++c) p += qreg[c] * kp[lane + 32 * c];
    for (int off = 16; off > 0; off >>= 1) p += __shfl_xor(p, off);
    float lg = p * SCALE_ - rec * INV_TEMP_;
    if (lane == t) logit = lg;
  }
  float mx = logit;
  for (int off = 16; off > 0; off >>= 1) mx = fmaxf(mx, __shfl_xor(mx, off));
  float e = __expf(logit - mx);
  float s = e;
  for (int off = 16; off > 0; off >>= 1) s += __shfl_xor(s, off);
  float wgt = e / s;

  float acc[8];
#pragma unroll
  for (int c = 0; c < 8; ++c) acc[c] = 0.f;
  for (int t = 0; t < 32; ++t) {
    float wt = __shfl(wgt, t);
    int j = __shfl(keepIdx, t);
    const float* vp = v + ((size_t)b * HW_ + j) * D_;
#pragma unroll
    for (int c = 0; c < 8; ++c) acc[c] += wt * vp[lane + 32 * c];
  }
  u16* frow = fused + (size_t)qi * D_;
#pragma unroll
  for (int c = 0; c < 8; ++c) frow[lane + 32 * c] = f2bf(acc[c]);
}

// ---------------------------------------------------------------- launch
extern "C" void kernel_launch(void* const* d_in, const int* in_sizes, int n_in,
                              void* d_out, int out_size, void* d_ws, size_t ws_size,
                              hipStream_t stream)
{
  const float* qmap = (const float*)d_in[0];
  const float* kmap = (const float*)d_in[1];
  const float* pq   = (const float*)d_in[2];
  const float* pk   = (const float*)d_in[3];
  const float* gq   = (const float*)d_in[4];
  const float* bqln = (const float*)d_in[5];
  const float* gk   = (const float*)d_in[6];
  const float* bkln = (const float*)d_in[7];
  const float* Wq   = (const float*)d_in[8];
  const float* bq   = (const float*)d_in[9];
  const float* Wk   = (const float*)d_in[10];
  const float* bk   = (const float*)d_in[11];
  const float* Wv   = (const float*)d_in[12];
  const float* bv   = (const float*)d_in[13];
  const float* Wo   = (const float*)d_in[14];
  const float* bo   = (const float*)d_in[15];
  float* out = (float*)d_out;

  char* wp = (char*)d_ws;
  auto alloc = [&](size_t bytes) -> void* {
    void* p = (void*)wp; wp += (bytes + 255) & ~(size_t)255; return p;
  };
  u16* qn    = (u16*)alloc((size_t)M_ * C_ * 2);
  u16* kn    = (u16*)alloc((size_t)M_ * C_ * 2);
  u16* vt    = (u16*)alloc((size_t)M_ * C_ * 2);
  u16* wq_bf = (u16*)alloc((size_t)D_ * C_ * 2);
  u16* wk_bf = (u16*)alloc((size_t)D_ * C_ * 2);
  u16* wv_bf = (u16*)alloc((size_t)D_ * C_ * 2);
  u16* wo_bf = (u16*)alloc((size_t)C_ * D_ * 2);
  float* qf  = (float*)alloc((size_t)M_ * D_ * 4);
  float* kf  = (float*)alloc((size_t)M_ * D_ * 4);
  float* vf  = (float*)alloc((size_t)M_ * D_ * 4);
  float* dqr = (float*)alloc((size_t)M_ * 3 * 4);
  float* mqr = (float*)alloc((size_t)M_ * 3 * 4);
  float* dkr = (float*)alloc((size_t)M_ * 3 * 4);
  float* mkr = (float*)alloc((size_t)M_ * 3 * 4);
  float* nqr = (float*)alloc((size_t)M_ * 4);
  float* nkr = (float*)alloc((size_t)M_ * 4);
  u16* fused = (u16*)alloc((size_t)M_ * D_ * 2);

  k_cvt_bf16<<<256, 256, 0, stream>>>(Wq, wq_bf, D_ * C_);
  k_cvt_bf16<<<256, 256, 0, stream>>>(Wk, wk_bf, D_ * C_);
  k_cvt_bf16<<<256, 256, 0, stream>>>(Wv, wv_bf, D_ * C_);
  k_cvt_bf16<<<256, 256, 0, stream>>>(Wo, wo_bf, C_ * D_);

  k_prep<<<B_ * (HW_ / 32), 256, 0, stream>>>(qmap, kmap, gq, bqln, gk, bkln, qn, kn, vt);
  k_rays<<<(M_ + 255) / 256, 256, 0, stream>>>(pq, pk, dqr, mqr, nqr, dkr, mkr, nkr);

  dim3 gp(M_ / 128, D_ / 64);
  k_gemm_wmma<<<gp, 256, 0, stream>>>(qn, wq_bf, bq, qf, C_, D_, 0);
  k_gemm_wmma<<<gp, 256, 0, stream>>>(kn, wk_bf, bk, kf, C_, D_, 0);
  k_gemm_wmma<<<gp, 256, 0, stream>>>(vt, wv_bf, bv, vf, C_, D_, 0);

  k_attn_topk<<<M_ / 8, 256, 0, stream>>>(qf, kf, vf, dqr, mqr, nqr, dkr, mkr, nkr, fused);

  dim3 go(M_ / 128, C_ / 64);
  k_gemm_wmma<<<go, 256, 0, stream>>>(fused, wo_bf, bo, out, D_, C_, 1);
}
